// TriBranchUVFusion_3324304687209
// MI455X (gfx1250) — matrix operand order repaired and checked by reading, not compiled, not run on hardware
//
#include <hip/hip_runtime.h>
#include <hip/hip_bf16.h>
#include <math.h>

typedef __attribute__((ext_vector_type(16))) _Float16 v16h;
typedef __attribute__((ext_vector_type(8)))  float    v8f;

// ---------------- problem constants ----------------
#define AT   256
#define PP   65536            // AT*AT
#define NC   64               // feature channels per branch
#define CO_  192              // refine output channels
#define C3_  194              // ctx channels
#define HID_ 48
#define CPG  24               // CO_/8 groups

// ---------------- workspace layout ----------------
#define F_REGION1   0
#define F_COUNTS    25165824              // 3*2*65536 floats
#define F_STATS     (25165824 + 393216)   // 1024 floats of small stats
#define F_ZERO_TOT  (25165824 + 393216 + 1024)
#define ST_GN1   0
#define ST_GN2   32
#define ST_MAX   64
#define ST_POOL  66
#define ST_GATE  450
#define ST_SCAL  834
#define B_CTX16  102240256ULL   // 2*194*65536 halves
#define B_H1F16  153096192ULL   // 2*192*65536 halves
#define B_W16    203427840ULL   // 704256 halves (c1w|c2w|rpw)
#define W16_C1   0
#define W16_C2   335232
#define W16_RP   667008
#define W16_TOT  704256
#define O_G      0
#define O_B      8388608
#define O_H      16777216
#define O_COV    25165824
#define O_UNC    25296896
#define O_CONF   25427968

__device__ __forceinline__ float sigmoidf_(float x){ return 1.f/(1.f+expf(-x)); }

// ---------------------------------------------------------------- zero
__global__ __launch_bounds__(256) void k_zero(float* ws, int n){
  int i = blockIdx.x*256 + threadIdx.x;
  if (i < n) ws[i] = 0.f;
}

// ---------------------------------------------------------------- cast weights to f16
__global__ __launch_bounds__(256) void k_cast_w(const float* __restrict__ c1w,
                                                const float* __restrict__ c2w,
                                                const float* __restrict__ rpw,
                                                _Float16* __restrict__ w16){
  int i = blockIdx.x*256 + threadIdx.x;
  if (i < W16_C2)            w16[i] = (_Float16)c1w[i];
  else if (i < W16_RP)       w16[i] = (_Float16)c2w[i - W16_C2];
  else if (i < W16_TOT)      w16[i] = (_Float16)rpw[i - W16_RP];
}

// ---------------------------------------------------------------- scatter fuse (3 branches)
__global__ __launch_bounds__(256) void k_scatter(const float* __restrict__ vf,
                                                 const float* __restrict__ uvs,
                                                 const float* __restrict__ masks,
                                                 const float* __restrict__ wg_,
                                                 const float* __restrict__ wb_,
                                                 const float* __restrict__ wh_,
                                                 float* __restrict__ atlas,
                                                 float* __restrict__ counts){
  int t = blockIdx.x*256 + threadIdx.x;           // B*V*H*W = 262144
  int bv  = t >> 14;
  int pix = t & 16383;
  float u = uvs[2*t], v = uvs[2*t+1];
  if (!(__builtin_isfinite(u) && __builtin_isfinite(v))) return;
  if (masks[t] <= 0.5f) return;
  float uc = fminf(fmaxf(u,0.f),1.f);
  float vc = fminf(fmaxf(v,0.f),1.f);
  int xi = (int)rintf(uc * 255.f);
  int yi = (int)rintf((1.f - vc) * 255.f);
  int idx = yi*AT + xi;
  int b = bv >> 3;
  float w3[3] = { fmaxf(wg_[bv],0.f), fmaxf(wb_[bv],0.f), fmaxf(wh_[bv],0.f) };
  #pragma unroll
  for (int j=0;j<3;j++)
    if (w3[j] > 0.f) atomicAdd(&counts[(j*2+b)*PP + idx], w3[j]);
  const float* src = vf + ((size_t)bv*NC << 14) + pix;
  for (int c=0;c<NC;c++){
    float f = src[(size_t)c << 14];
    #pragma unroll
    for (int j=0;j<3;j++)
      if (w3[j] > 0.f)
        atomicAdd(&atlas[(((j*2+b)*NC + c) << 16) + idx], w3[j]*f);
  }
}

// ---------------------------------------------------------------- fused feats + coverage + raw uncertainty
__global__ __launch_bounds__(256) void k_fuse(const float* __restrict__ atlas,
                                              const float* __restrict__ counts,
                                              float* __restrict__ out,
                                              _Float16* __restrict__ ctx16,
                                              float* __restrict__ stats){
  int t = blockIdx.x*256 + threadIdx.x;           // 131072
  int b = t >> 16, p = t & 65535;
  float cg = counts[(0+b)*PP + p];
  float cb = counts[(2+b)*PP + p];
  float ch = counts[(4+b)*PP + p];
  float nv = (cg>0.f?1.f:0.f)+(cb>0.f?1.f:0.f)+(ch>0.f?1.f:0.f);
  float cov = fminf(nv*(1.f/3.f), 1.f);
  out[O_COV + t] = cov;
  ctx16[((b*C3_ + 192) << 16) + p] = (_Float16)cov;
  float ig = 1.f/fmaxf(cg,1.f), ib = 1.f/fmaxf(cb,1.f), ih = 1.f/fmaxf(ch,1.f);
  float accv = 0.f;
  for (int c=0;c<NC;c++){
    float g  = atlas[(((0+b)*NC + c) << 16) + p] * ig;
    float bb = atlas[(((2+b)*NC + c) << 16) + p] * ib;
    float hh = atlas[(((4+b)*NC + c) << 16) + p] * ih;
    int o = ((b*NC + c) << 16) + p;
    out[O_G + o] = g; out[O_B + o] = bb; out[O_H + o] = hh;
    ctx16[((b*C3_ + c      ) << 16) + p] = (_Float16)g;
    ctx16[((b*C3_ + 64  + c) << 16) + p] = (_Float16)bb;
    ctx16[((b*C3_ + 128 + c) << 16) + p] = (_Float16)hh;
    float mu = (g+bb+hh)*(1.f/3.f);
    float dg=g-mu, db=bb-mu, dh=hh-mu;
    accv += (dg*dg + db*db + dh*dh)*(1.f/3.f);
  }
  float ur = sqrtf(accv*(1.f/NC));
  out[O_UNC + t] = ur;
  __shared__ float sm[256];
  sm[threadIdx.x] = ur; __syncthreads();
  for (int s=128;s>0;s>>=1){
    if (threadIdx.x < s) sm[threadIdx.x] = fmaxf(sm[threadIdx.x], sm[threadIdx.x+s]);
    __syncthreads();
  }
  if (threadIdx.x == 0)
    atomicMax((int*)(stats + ST_MAX + b), __float_as_int(sm[0]));
}

// ---------------------------------------------------------------- uncertainty normalize
__global__ __launch_bounds__(256) void k_unc(float* __restrict__ out,
                                             _Float16* __restrict__ ctx16,
                                             const float* __restrict__ stats){
  int t = blockIdx.x*256 + threadIdx.x;           // 131072
  int b = t >> 16, p = t & 65535;
  float scale = fmaxf(stats[ST_MAX + b], 1e-6f);
  float un = fminf(fmaxf(out[O_UNC + t]/scale, 0.f), 1.f) * out[O_COV + t];
  out[O_UNC + t] = un;
  ctx16[((b*C3_ + 193) << 16) + p] = (_Float16)un;
}

// ---------------------------------------------------------------- WMMA implicit-GEMM conv (KHxKH, SAME)
// block = 256 thr (8 wave32); block tile M=64 x N=128; K staged 32-wide through LDS.
// LDS tiles are stored PRE-PERMUTED into the WMMA per-lane fragment layout so each
// lane loads its v16h fragment as two ds_load_b128.
// A layout (16x32 f16, ISA 7.12.2): lane = (m&15) | ((k>>3)&1)<<4 ; slot = ((k&16)>>1)|(k&7)
// B layout (32x16 f16): lane = (n&15) | ((k>>3)&1)<<4 ; slot = ((k&16)>>1)|(k&7)
template<int KH>
__global__ __launch_bounds__(256) void k_gemm(const _Float16* __restrict__ in,
                                              const _Float16* __restrict__ wgt,   // [192, Cin*KH*KH]
                                              const float* __restrict__ bias,
                                              float* __restrict__ out,            // [2,192,256,256]
                                              int Cin,
                                              const float* __restrict__ alpha_ptr,
                                              int alpha_is_tanh, int beta){
  constexpr int KHW = KH*KH;
  constexpr int OFF = KH >> 1;
  const int Ktot = Cin*KHW;
  __shared__ __align__(32) _Float16 sA[4*32*16];   // [mtile][lane][slot]
  __shared__ __align__(32) _Float16 sB[8*32*16];   // [ntile][lane][slot]
  __shared__ int srow_off[32];
  __shared__ int srow_dx[32];
  const int tid  = threadIdx.x;
  const int lane = tid & 31;
  const int wave = tid >> 5;
  const int wm = wave >> 1;            // 0..3 -> M subtile
  const int wn = wave & 1;             // 0..1 -> N half (4 tiles each)
  const int mbase = blockIdx.y * 64;
  const int n0 = blockIdx.x * 128;     // 128 pixels: same (batch,row) always
  const int bb = n0 >> 16;
  const int y0 = (n0 & 65535) >> 8;
  const int x0 = n0 & 255;             // 0 or 128
  v8f acc[4] = {{},{},{},{}};
  for (int kk0 = 0; kk0 < Ktot; kk0 += 32){
    // per-k-row descriptors: channel/row base offset + dx (hoists div/mod + y-bounds)
    if (tid < 32){
      int kk  = kk0 + tid;
      int cin = kk / KHW;
      int rr  = kk - cin*KHW;
      int dy  = rr / KH - OFF;
      int dx  = rr % KH - OFF;
      int ye  = y0 + dy;
      bool ok = (kk < Ktot) && ((unsigned)ye < 256u);
      srow_off[tid] = ok ? (((bb*Cin + cin) << 16) + (ye << 8)) : -1;
      srow_dx[tid]  = dx;
    }
    __syncthreads();
    // stage A tile (weights, contiguous in K), permuted store
    #pragma unroll
    for (int i = 0; i < 8; i++){
      int e = tid + i*256;             // 0..2047
      int m = e >> 5, k = e & 31;
      int kk = kk0 + k;
      _Float16 v = (kk < Ktot) ? wgt[(mbase+m)*Ktot + kk] : (_Float16)0.f;
      int ln = (m & 15) | (((k >> 3) & 1) << 4);
      int sl = ((k & 16) >> 1) | (k & 7);
      sA[(((m >> 4)*32 + ln) << 4) | sl] = v;
    }
    // stage B tile (im2col gather), permuted store
    #pragma unroll
    for (int i = 0; i < 16; i++){
      int e = tid + i*256;             // 0..4095
      int row = e >> 7, nl = e & 127;
      int off = srow_off[row];
      int x = x0 + nl + srow_dx[row];
      _Float16 v = (off >= 0 && (unsigned)x < 256u) ? in[off + x] : (_Float16)0.f;
      int ln = (nl & 15) | (((row >> 3) & 1) << 4);
      int sl = ((row & 16) >> 1) | (row & 7);
      sB[(((nl >> 4)*32 + ln) << 4) | sl] = v;
    }
    __syncthreads();
    // fragments: one v16h A (reused 4x), four v16h B, four WMMAs
    v16h af = *(const v16h*)(sA + ((wm*32 + lane) << 4));
    #pragma unroll
    for (int j = 0; j < 4; j++){
      v16h bf = *(const v16h*)(sB + (((wn*4 + j)*32 + lane) << 4));
      acc[j] = __builtin_amdgcn_wmma_f32_16x16x32_f16(false, af, false, bf,
                                                      (short)0, acc[j], false, false);
    }
    __syncthreads();
  }
  float alpha = 1.f;
  if (alpha_ptr){ alpha = alpha_ptr[0]; if (alpha_is_tanh) alpha = tanhf(alpha); }
  const int lhp = lane & 15, lhi = lane >> 4;
  #pragma unroll
  for (int j = 0; j < 4; j++){
    int x = x0 + wn*64 + j*16 + lhp;
    #pragma unroll
    for (int r = 0; r < 8; r++){
      int m = mbase + wm*16 + r + lhi*8;
      int o = ((bb*CO_ + m) << 16) + (y0 << 8) + x;
      float prev = beta ? out[o] : 0.f;
      out[o] = prev + alpha*(acc[j][r] + bias[m]);
    }
  }
}

// ---------------------------------------------------------------- group-norm reductions
__global__ __launch_bounds__(256) void k_gnreduce(const float* __restrict__ x,
                                                  float* __restrict__ stats, int statoff){
  int bg = blockIdx.x;                 // 0..15  (b*8+g)
  int chan = blockIdx.y;               // 0..23
  int b = bg >> 3, g = bg & 7;
  int m = g*CPG + chan;
  const float* p = x + ((size_t)(b*CO_ + m) << 16);
  float s = 0.f, s2 = 0.f;
  for (int i = threadIdx.x; i < PP; i += 256){ float v = p[i]; s += v; s2 += v*v; }
  __shared__ float sh[512];
  sh[threadIdx.x] = s; sh[256+threadIdx.x] = s2; __syncthreads();
  for (int st=128; st>0; st>>=1){
    if (threadIdx.x < st){ sh[threadIdx.x] += sh[threadIdx.x+st];
                           sh[256+threadIdx.x] += sh[256+threadIdx.x+st]; }
    __syncthreads();
  }
  if (threadIdx.x == 0){
    atomicAdd(&stats[statoff + bg*2],     sh[0]);
    atomicAdd(&stats[statoff + bg*2 + 1], sh[256]);
  }
}

// ---------------------------------------------------------------- GN1 apply + silu -> f16
__global__ __launch_bounds__(256) void k_gn1(const float* __restrict__ x,
                                             const float* __restrict__ stats,
                                             const float* __restrict__ sc,
                                             const float* __restrict__ bi,
                                             _Float16* __restrict__ h16){
  int e = blockIdx.x*256 + threadIdx.x;        // 25165824
  int cidx = e >> 16;
  int b = cidx / CO_, m = cidx % CO_;
  int bg = b*8 + m/CPG;
  const float N = (float)(CPG*PP);
  float mean = stats[ST_GN1 + bg*2] / N;
  float var  = stats[ST_GN1 + bg*2 + 1] / N - mean*mean;
  float v = (x[e]-mean)*rsqrtf(var + 1e-5f)*sc[m] + bi[m];
  v = v * sigmoidf_(v);
  h16[e] = (_Float16)v;
}

// ---------------------------------------------------------------- GN2 apply + silu in place + channel pool
__global__ __launch_bounds__(256) void k_gn2(float* __restrict__ x,
                                             float* __restrict__ stats,
                                             const float* __restrict__ sc,
                                             const float* __restrict__ bi){
  int e = blockIdx.x*256 + threadIdx.x;
  int cidx = e >> 16;
  int b = cidx / CO_, m = cidx % CO_;
  int bg = b*8 + m/CPG;
  const float N = (float)(CPG*PP);
  float mean = stats[ST_GN2 + bg*2] / N;
  float var  = stats[ST_GN2 + bg*2 + 1] / N - mean*mean;
  float v = (x[e]-mean)*rsqrtf(var + 1e-5f)*sc[m] + bi[m];
  v = v * sigmoidf_(v);
  x[e] = v;
  __shared__ float sh[256];
  sh[threadIdx.x] = v; __syncthreads();
  for (int s=128;s>0;s>>=1){
    if (threadIdx.x < s) sh[threadIdx.x] += sh[threadIdx.x+s];
    __syncthreads();
  }
  if (threadIdx.x == 0) atomicAdd(&stats[ST_POOL + cidx], sh[0]);
}

// ---------------------------------------------------------------- channel-gate MLP + scalars (1 block)
__global__ __launch_bounds__(256) void k_gate(const float* __restrict__ gw1,
                                              const float* __restrict__ gb1,
                                              const float* __restrict__ gw2,
                                              const float* __restrict__ gb2,
                                              const float* __restrict__ rs,
                                              const float* __restrict__ mix,
                                              float* __restrict__ stats){
  __shared__ float sp[384];
  __shared__ float sh[96];
  int t = threadIdx.x;
  for (int i=t;i<384;i+=256) sp[i] = stats[ST_POOL + i] * (1.f/65536.f);
  __syncthreads();
  for (int i=t;i<96;i+=256){
    int b = i/HID_, h = i%HID_;
    float a = gb1[h];
    for (int c=0;c<CO_;c++) a += sp[b*CO_ + c]*gw1[h*CO_ + c];
    sh[i] = a * sigmoidf_(a);
  }
  __syncthreads();
  for (int i=t;i<384;i+=256){
    int b = i/CO_, c = i%CO_;
    float a = gb2[c];
    for (int h=0;h<HID_;h++) a += sh[b*HID_ + h]*gw2[c*HID_ + h];
    stats[ST_GATE + i] = sigmoidf_(a);
  }
  if (t == 0){ stats[ST_SCAL] = tanhf(rs[0]); stats[ST_SCAL+1] = tanhf(mix[0]); }
}

// ---------------------------------------------------------------- apply gate in place
__global__ __launch_bounds__(256) void k_gateapply(float* __restrict__ x,
                                                   const float* __restrict__ stats){
  int e = blockIdx.x*256 + threadIdx.x;
  int cidx = e >> 16;
  x[e] *= 0.5f + stats[ST_GATE + cidx];
}

// ---------------------------------------------------------------- confidence (1x1, 3 outputs)
__global__ __launch_bounds__(256) void k_conf(const _Float16* __restrict__ ctx16,
                                              const float* __restrict__ cw,
                                              const float* __restrict__ cb,
                                              float* __restrict__ out){
  int t = blockIdx.x*256 + threadIdx.x;        // 131072
  int b = t >> 16, p = t & 65535;
  float a0 = cb[0], a1 = cb[1], a2 = cb[2];
  for (int c=0;c<C3_;c++){
    float v = (float)ctx16[((b*C3_ + c) << 16) + p];
    a0 += v*cw[c]; a1 += v*cw[C3_ + c]; a2 += v*cw[2*C3_ + c];
  }
  out[O_CONF + ((b*3 + 0) << 16) + p] = sigmoidf_(a0);
  out[O_CONF + ((b*3 + 1) << 16) + p] = sigmoidf_(a1);
  out[O_CONF + ((b*3 + 2) << 16) + p] = sigmoidf_(a2);
}

// ---------------------------------------------------------------- final mix: out_j += ms*refine*conf_j
__global__ __launch_bounds__(256) void k_final(const float* __restrict__ refine,
                                               const float* __restrict__ stats,
                                               float* __restrict__ out){
  int e = blockIdx.x*256 + threadIdx.x;        // 25165824
  int cidx = e >> 16, p = e & 65535;
  int b = cidx / CO_, m = cidx % CO_;
  int j = m >> 6, c = m & 63;
  float confv = out[O_CONF + ((b*3 + j) << 16) + p];
  float ms = stats[ST_SCAL + 1];
  out[j*8388608 + ((b*NC + c) << 16) + p] += ms * refine[e] * confv;
}

// ================================================================ launch
extern "C" void kernel_launch(void* const* d_in, const int* in_sizes, int n_in,
                              void* d_out, int out_size, void* d_ws, size_t ws_size,
                              hipStream_t stream){
  const float* vf    = (const float*)d_in[0];
  const float* uvs   = (const float*)d_in[1];
  const float* masks = (const float*)d_in[2];
  const float* wg    = (const float*)d_in[3];
  const float* wb    = (const float*)d_in[4];
  const float* wh    = (const float*)d_in[5];
  const float* c1w   = (const float*)d_in[7];
  const float* c1b   = (const float*)d_in[8];
  const float* g1s   = (const float*)d_in[9];
  const float* g1b   = (const float*)d_in[10];
  const float* c2w   = (const float*)d_in[11];
  const float* c2b   = (const float*)d_in[12];
  const float* g2s   = (const float*)d_in[13];
  const float* g2b   = (const float*)d_in[14];
  const float* gw1   = (const float*)d_in[15];
  const float* gb1   = (const float*)d_in[16];
  const float* gw2   = (const float*)d_in[17];
  const float* gb2   = (const float*)d_in[18];
  const float* rpb   = (const float*)d_in[20];
  const float* rs    = (const float*)d_in[21];
  const float* cw    = (const float*)d_in[22];
  const float* cb    = (const float*)d_in[23];
  const float* mix   = (const float*)d_in[24];

  float* out = (float*)d_out;
  float* wsf     = (float*)d_ws;
  float* region1 = wsf + F_REGION1;
  float* counts  = wsf + F_COUNTS;
  float* stats   = wsf + F_STATS;
  _Float16* ctx16 = (_Float16*)((char*)d_ws + B_CTX16);
  _Float16* h1f16 = (_Float16*)((char*)d_ws + B_H1F16);
  _Float16* w16   = (_Float16*)((char*)d_ws + B_W16);
  _Float16* c1w16 = w16 + W16_C1;
  _Float16* c2w16 = w16 + W16_C2;
  _Float16* rpw16 = w16 + W16_RP;

  // 0) zero accumulators + stats
  k_zero<<<(F_ZERO_TOT + 255)/256, 256, 0, stream>>>(wsf, F_ZERO_TOT);
  // 1) weights -> f16
  k_cast_w<<<(W16_TOT + 255)/256, 256, 0, stream>>>(c1w, c2w, (const float*)d_in[19], w16);
  // 2) scatter fuse (3 branches) -- atomics stay in 192MB L2 (atlas ~100MB)
  k_scatter<<<262144/256, 256, 0, stream>>>(vf, uvs, masks, wg, wb, wh, region1, counts);
  // 3) fused feats + coverage + raw uncertainty (+per-sample max)
  k_fuse<<<131072/256, 256, 0, stream>>>(region1, counts, out, ctx16, stats);
  // 4) uncertainty normalize
  k_unc<<<131072/256, 256, 0, stream>>>(out, ctx16, stats);
  // 5) confidence head
  k_conf<<<131072/256, 256, 0, stream>>>(ctx16, cw, cb, out);
  // 6) conv1 (194 -> 192, 3x3) WMMA
  k_gemm<3><<<dim3(1024,3), 256, 0, stream>>>(ctx16, c1w16, c1b, region1,
                                              C3_, nullptr, 0, 0);
  // 7) GN1
  k_gnreduce<<<dim3(16,24), 256, 0, stream>>>(region1, stats, ST_GN1);
  k_gn1<<<25165824/256, 256, 0, stream>>>(region1, stats, g1s, g1b, h1f16);
  // 8) conv2 (192 -> 192, 3x3) WMMA
  k_gemm<3><<<dim3(1024,3), 256, 0, stream>>>(h1f16, c2w16, c2b, region1,
                                              CO_, nullptr, 0, 0);
  // 9) GN2 + silu in place + channel pool
  k_gnreduce<<<dim3(16,24), 256, 0, stream>>>(region1, stats, ST_GN2);
  k_gn2<<<25165824/256, 256, 0, stream>>>(region1, stats, g2s, g2b);
  // 10) channel gate MLP + scalars
  k_gate<<<1, 256, 0, stream>>>(gw1, gb1, gw2, gb2, rs, mix, stats);
  k_gateapply<<<25165824/256, 256, 0, stream>>>(region1, stats);
  // 11) residual 1x1 GEMM accumulated: region1 += tanh(rs)*(ctx @ rpw^T + rpb)
  k_gemm<1><<<dim3(1024,3), 256, 0, stream>>>(ctx16, rpw16, rpb, region1,
                                              C3_, rs, 1, 1);
  // 12) final mix into branch outputs
  k_final<<<25165824/256, 256, 0, stream>>>(region1, stats, out);
}
